// STSN_48361331753354
// MI455X (gfx1250) — matrix operand order repaired and checked by reading, not compiled, not run on hardware
//
#include <hip/hip_runtime.h>
#include <hip/hip_bf16.h>
#include <stdint.h>

typedef __attribute__((ext_vector_type(16))) _Float16 v16h;
typedef __attribute__((ext_vector_type(8)))  _Float16 v8h;
typedef __attribute__((ext_vector_type(8)))  float    v8f;
typedef __attribute__((ext_vector_type(4)))  float    v4f;

static constexpr int Bm = 32, Hh = 80, Ww = 80, Ss = 9, ITERS = 3;
static constexpr int HWp = Hh * Ww;            // 6400
static constexpr int BSn = Bm * Ss;            // 288
__device__ static constexpr float SCALEc = 0.17677669529663687f; // 32^-0.5

#define SHUF16(a, b) __builtin_shufflevector(a, b, 0,1,2,3,4,5,6,7,8,9,10,11,12,13,14,15)

// ---------------------------------------------------------------------------
// WMMA 16-bit operand lane layouts (ISA 7.12.2):
//   A (16x32): lane<16 -> M=lane, K in {0..7,16..23}; lane>=16 -> K+8
//   B (32x16): lane<16 -> N=lane, K in {0..7,16..23}; lane>=16 -> K+8
// Weight tables are pre-packed fragment-major: [frag][lane][16 halves],
// so every lane reads its B-fragment as two contiguous 16B chunks.
// ---------------------------------------------------------------------------

// Conv 5x5 weights -> fragment-major f16: wf[((t*2+ntile)*32+lane)*16 + i]
__global__ void pack_conv5_frag(const float* __restrict__ w, _Float16* __restrict__ wf,
                                int iohw_flip) {
  int idx = blockIdx.x * blockDim.x + threadIdx.x;
  if (idx >= 25 * 2 * 32 * 16) return;
  int i = idx & 15;
  int lane = (idx >> 4) & 31;
  int ntile = (idx >> 9) & 1;
  int t = idx >> 10;                       // dy*5+dx of the equivalent conv
  int hi = lane >> 4;
  int co = ntile * 16 + (lane & 15);
  int ci = i + (i & 8) + (hi ? 8 : 0);
  int dy = t / 5, dx = t - dy * 5;
  float v;
  if (iohw_flip)  v = w[(ci * 32 + co) * 25 + (4 - dy) * 5 + (4 - dx)]; // convT IOHW
  else            v = w[(co * 32 + ci) * 25 + t];                        // conv  OIHW
  wf[idx] = (_Float16)v;
}

// Linear weights w(N,32) -> fragment-major: wf[((nt*32)+lane)*16 + i]
__global__ void pack_lin_frag(const float* __restrict__ w, _Float16* __restrict__ wf, int N) {
  int idx = blockIdx.x * blockDim.x + threadIdx.x;
  if (idx >= N * 32) return;               // (N/16)*32*16
  int i = idx & 15;
  int lane = (idx >> 4) & 31;
  int nt = idx >> 9;
  int hi = lane >> 4;
  int n = nt * 16 + (lane & 15);
  int k = i + (i & 8) + (hi ? 8 : 0);
  wf[idx] = (_Float16)w[n * 32 + k];
}

// pe[p*32+c] = b[c] + grid4(p) . w[c][0..3]
__global__ void pe_kernel(const float* __restrict__ w, const float* __restrict__ b,
                          float* __restrict__ pe) {
  int idx = blockIdx.x * blockDim.x + threadIdx.x;
  if (idx >= HWp * 32) return;
  int p = idx >> 5, c = idx & 31;
  int i = p / Ww, j = p - i * Ww;
  float r0 = (float)i * (1.0f / 79.0f);
  float r1 = (float)j * (1.0f / 79.0f);
  pe[idx] = b[c] + r0 * w[c * 4 + 0] + r1 * w[c * 4 + 1]
                 + (1.0f - r0) * w[c * 4 + 2] + (1.0f - r1) * w[c * 4 + 3];
}

// Encoder conv1: Cin=1, 5x5 pad2, relu; fp32 NCHW in -> f16 NHWC out
__global__ void conv1_kernel(const float* __restrict__ x, const float* __restrict__ w,
                             const float* __restrict__ bias, _Float16* __restrict__ out) {
  int idx = blockIdx.x * blockDim.x + threadIdx.x;      // ((b*HW+p)*32+co)
  if (idx >= Bm * HWp * 32) return;
  int co = idx & 31;
  int rest = idx >> 5;
  int p = rest % HWp;
  int b = rest / HWp;
  int y = p / Ww, xx = p - y * Ww;
  const float* xb = x + (size_t)b * HWp;
  float s = bias[co];
  for (int dy = 0; dy < 5; ++dy) {
    int yg = y + dy - 2;
    if (yg < 0 || yg >= Hh) continue;
    for (int dx = 0; dx < 5; ++dx) {
      int xg = xx + dx - 2;
      if (xg < 0 || xg >= Ww) continue;
      s += xb[yg * Ww + xg] * w[co * 25 + dy * 5 + dx];
    }
  }
  out[idx] = (_Float16)fmaxf(s, 0.0f);
}

// ---------------------------------------------------------------------------
// 5x5 pad2 32->32 conv, f16 NHWC in/out, implicit-GEMM WMMA.
// WG = one output row; 10 waves = 5 pixel tiles x 2 Cout tiles.
// LDS tile [5][84][40] halves (stride-40 padding => bank spread, 16B aligned
// chunks); A-fragment = 2x ds b128, B-fragment = 2x global b128.
// ---------------------------------------------------------------------------
__global__ __launch_bounds__(320) void conv5_wmma(const _Float16* __restrict__ in,
                                                  const _Float16* __restrict__ wf,
                                                  const float* __restrict__ bias,
                                                  _Float16* __restrict__ out, int relu) {
  __shared__ __align__(16) _Float16 s_in[5 * 84 * 40];   // 33600 B
  const int img = blockIdx.x / Hh;
  const int y = blockIdx.x % Hh;

  __builtin_prefetch(wf, 0, 1);                          // global_prefetch_b8

  // zero-fill (halo columns / out-of-range rows stay zero)
  {
    v8h z = {};
    for (int ch = threadIdx.x; ch < (5 * 84 * 40) / 8; ch += 320)
      *(v8h*)&s_in[ch * 8] = z;
  }
  __syncthreads();
  // copy 5 input rows; one row = 80*32 halves = 320 threads x 16B
  for (int r = 0; r < 5; ++r) {
    int yg = y + r - 2;
    if (yg < 0 || yg >= Hh) continue;                    // block-uniform
    const _Float16* src = in + ((size_t)img * HWp + (size_t)yg * Ww) * 32;
    int t = threadIdx.x;
    int xg = t >> 2, ci = (t & 3) * 8;
    *(v8h*)&s_in[((r * 84) + (xg + 2)) * 40 + ci] = *(const v8h*)&src[t * 8];
  }
  __syncthreads();

  const int wave = threadIdx.x >> 5;
  const int lane = threadIdx.x & 31;
  const int ptile = wave % 5;                            // x0 = ptile*16
  const int ntile = wave / 5;                            // 0..1
  const int x0 = ptile * 16;
  const int ln = lane & 15;
  const int hi = (lane & 16) ? 1 : 0;
  const int koff = hi ? 8 : 0;

  v8f acc = {};
  for (int dy = 0; dy < 5; ++dy) {
    for (int dx = 0; dx < 5; ++dx) {
      const _Float16* ap = &s_in[(dy * 84 + (x0 + ln + dx)) * 40 + koff];
      v8h a0 = *(const v8h*)ap;                          // K = koff..koff+7
      v8h a1 = *(const v8h*)(ap + 16);                   // K = 16+koff..+7
      const _Float16* bp = &wf[(((size_t)(dy * 5 + dx) * 2 + ntile) * 32 + lane) * 16];
      v8h b0 = *(const v8h*)bp;
      v8h b1 = *(const v8h*)(bp + 8);
      v16h af = SHUF16(a0, a1);
      v16h bf = SHUF16(b0, b1);
      acc = __builtin_amdgcn_wmma_f32_16x16x32_f16(false, af, false, bf,
                                                   (short)0, acc, false, false);
    }
  }
#pragma unroll
  for (int r = 0; r < 8; ++r) {
    int mm = r + (hi ? 8 : 0);                           // pixel within tile
    int co = ntile * 16 + ln;                            // out channel
    float v = acc[r] + bias[co];
    if (relu) v = fmaxf(v, 0.0f);
    out[((size_t)img * HWp + (size_t)y * Ww + (x0 + mm)) * 32 + co] = (_Float16)v;
  }
}

// ---------------------------------------------------------------------------
// Y[M][N] = act(X[M][32] . W^T + b); K=32 => one WMMA per 16x16 tile.
// A: four 16B v4f global loads per lane; B: fragment-major f16 table.
// ---------------------------------------------------------------------------
__global__ __launch_bounds__(256) void lin32_wmma(const float* __restrict__ X,
                                                  const _Float16* __restrict__ Wf,
                                                  const float* __restrict__ bias,
                                                  float* __restrict__ Y,
                                                  int M, int N, int relu) {
  const int wavesPerBlk = blockDim.x >> 5;
  const int mt = blockIdx.x * wavesPerBlk + (threadIdx.x >> 5);
  if (mt * 16 >= M) return;                              // wave-uniform exit
  const int lane = threadIdx.x & 31;
  const int ln = lane & 15;
  const int hi = (lane & 16) ? 1 : 0;
  const int row = mt * 16 + ln;
  const float* xr = X + (size_t)row * 32 + (hi ? 8 : 0);
  v4f c0 = *(const v4f*)(xr + 0);
  v4f c1 = *(const v4f*)(xr + 4);
  v4f c2 = *(const v4f*)(xr + 16);
  v4f c3 = *(const v4f*)(xr + 20);
  v16h af;
#pragma unroll
  for (int i = 0; i < 4; ++i) {
    af[i]      = (_Float16)c0[i];
    af[4 + i]  = (_Float16)c1[i];
    af[8 + i]  = (_Float16)c2[i];
    af[12 + i] = (_Float16)c3[i];
  }
  for (int nt = 0; nt < N / 16; ++nt) {
    const _Float16* bp = Wf + ((size_t)nt * 32 + lane) * 16;
    v8h b0 = *(const v8h*)bp;
    v8h b1 = *(const v8h*)(bp + 8);
    v16h bf = SHUF16(b0, b1);
    v8f c = {};
    c = __builtin_amdgcn_wmma_f32_16x16x32_f16(false, af, false, bf,
                                               (short)0, c, false, false);
#pragma unroll
    for (int r = 0; r < 8; ++r) {
      int mm = r + (hi ? 8 : 0);
      int co = nt * 16 + ln;
      float v = c[r] + bias[co];
      if (relu) v = fmaxf(v, 0.0f);
      Y[(size_t)(mt * 16 + mm) * N + co] = v;
    }
  }
}

// h16 (B,HW,32 f16 NHWC) + pe -> LayerNorm over whole (HW,C) slab -> f32
__global__ __launch_bounds__(256) void enc_addpe_ln(const _Float16* __restrict__ h16,
                                                    const float* __restrict__ pe,
                                                    const float* __restrict__ g,
                                                    const float* __restrict__ bta,
                                                    float* __restrict__ out) {
  __shared__ float s_sum[256], s_sq[256];
  const int b = blockIdx.x;
  const size_t base = (size_t)b * 32 * HWp;
  float s = 0.0f, q = 0.0f;
  for (int e = threadIdx.x; e < 32 * HWp; e += 256) {
    float v = (float)h16[base + e] + pe[e];
    s += v; q += v * v;
  }
  s_sum[threadIdx.x] = s; s_sq[threadIdx.x] = q;
  __syncthreads();
  for (int off = 128; off > 0; off >>= 1) {
    if (threadIdx.x < off) { s_sum[threadIdx.x] += s_sum[threadIdx.x + off];
                             s_sq[threadIdx.x]  += s_sq[threadIdx.x + off]; }
    __syncthreads();
  }
  const float inv_n = 1.0f / (32.0f * HWp);
  const float mean = s_sum[0] * inv_n;
  const float var  = s_sq[0] * inv_n - mean * mean;
  const float inv  = rsqrtf(var + 1e-5f);
  for (int e = threadIdx.x; e < 32 * HWp; e += 256) {
    float v = (float)h16[base + e] + pe[e];
    out[base + e] = (v - mean) * inv * g[e] + bta[e];
  }
}

// Per-row LayerNorm over 32 channels
__global__ void ln_rows(const float* __restrict__ X, const float* __restrict__ g,
                        const float* __restrict__ bta, float* __restrict__ Y, int M) {
  int row = blockIdx.x * blockDim.x + threadIdx.x;
  if (row >= M) return;
  const float* x = X + (size_t)row * 32;
  float s = 0.0f;
  for (int c = 0; c < 32; ++c) s += x[c];
  float m = s * (1.0f / 32.0f);
  float q = 0.0f;
  for (int c = 0; c < 32; ++c) { float d = x[c] - m; q += d * d; }
  float inv = rsqrtf(q * (1.0f / 32.0f) + 1e-5f);
  float* y = Y + (size_t)row * 32;
  for (int c = 0; c < 32; ++c) y[c] = (x[c] - m) * inv * g[c] + bta[c];
}

__global__ void slots_init(const float* __restrict__ mu, const float* __restrict__ sigma,
                           const float* __restrict__ noise, float* __restrict__ slots) {
  int idx = blockIdx.x * blockDim.x + threadIdx.x;
  if (idx >= BSn * 32) return;
  int c = idx & 31;
  slots[idx] = mu[c] + sigma[c] * noise[idx];
}

__global__ void copyf(const float* __restrict__ src, float* __restrict__ dst, int n) {
  int i = blockIdx.x * blockDim.x + threadIdx.x;
  if (i < n) dst[i] = src[i];
}
__global__ void addf(float* __restrict__ dst, const float* __restrict__ src, int n) {
  int i = blockIdx.x * blockDim.x + threadIdx.x;
  if (i < n) dst[i] += src[i];
}

// dots[b,i,j] = SCALE * <q[b,i,:], keys[b,j,:]>
__global__ void dots_kernel(const float* __restrict__ q, const float* __restrict__ keys,
                            float* __restrict__ dots) {
  int idx = blockIdx.x * blockDim.x + threadIdx.x;       // b*HW + j
  if (idx >= Bm * HWp) return;
  int b = idx / HWp, j = idx - b * HWp;
  const float* kr = keys + (size_t)idx * 32;
  float kv[32];
  for (int c = 0; c < 32; ++c) kv[c] = kr[c];
  for (int i = 0; i < Ss; ++i) {
    const float* qr = q + (size_t)(b * Ss + i) * 32;
    float s = 0.0f;
    for (int c = 0; c < 32; ++c) s += qr[c] * kv[c];
    dots[(size_t)(b * Ss + i) * HWp + j] = s * SCALEc;
  }
}

// softmax over slot axis (per b,j) then +1e-8
__global__ void softmax_slots(const float* __restrict__ dots, float* __restrict__ attn) {
  int idx = blockIdx.x * blockDim.x + threadIdx.x;
  if (idx >= Bm * HWp) return;
  int b = idx / HWp, j = idx - b * HWp;
  float v[Ss], mx = -3.4e38f;
  for (int i = 0; i < Ss; ++i) {
    v[i] = dots[(size_t)(b * Ss + i) * HWp + j];
    mx = fmaxf(mx, v[i]);
  }
  float sum = 0.0f;
  for (int i = 0; i < Ss; ++i) { v[i] = expf(v[i] - mx); sum += v[i]; }
  float inv = 1.0f / sum;
  for (int i = 0; i < Ss; ++i)
    attn[(size_t)(b * Ss + i) * HWp + j] = v[i] * inv + 1e-8f;
}

// renormalize each (b,i) row over j
__global__ __launch_bounds__(256) void attn_renorm(float* __restrict__ attn) {
  __shared__ float red[256];
  float* a = attn + (size_t)blockIdx.x * HWp;
  float s = 0.0f;
  for (int j = threadIdx.x; j < HWp; j += 256) s += a[j];
  red[threadIdx.x] = s; __syncthreads();
  for (int off = 128; off > 0; off >>= 1) {
    if (threadIdx.x < off) red[threadIdx.x] += red[threadIdx.x + off];
    __syncthreads();
  }
  float inv = 1.0f / red[0];
  for (int j = threadIdx.x; j < HWp; j += 256) a[j] *= inv;
}

// upd[b,i,c] = sum_j values[b,j,c]*attn[b,i,j]
__global__ __launch_bounds__(256) void upd_kernel(const float* __restrict__ values,
                                                  const float* __restrict__ attn,
                                                  float* __restrict__ upd) {
  __shared__ float red[256];
  int bi = blockIdx.x;
  int b = bi / Ss;
  int c = threadIdx.x & 31;
  int jg = threadIdx.x >> 5;
  const float* at = attn + (size_t)bi * HWp;
  const float* vb = values + (size_t)b * HWp * 32;
  float s = 0.0f;
  for (int j = jg; j < HWp; j += 8) s += vb[(size_t)j * 32 + c] * at[j];
  red[threadIdx.x] = s; __syncthreads();
  if (threadIdx.x < 32) {
    float t = 0.0f;
    for (int g2 = 0; g2 < 8; ++g2) t += red[g2 * 32 + c];
    upd[(size_t)bi * 32 + c] = t;
  }
}

__global__ void gru_kernel(const float* __restrict__ gx, const float* __restrict__ gh,
                           const float* __restrict__ prev, float* __restrict__ slots) {
  int idx = blockIdx.x * blockDim.x + threadIdx.x;
  if (idx >= BSn * 32) return;
  int row = idx >> 5, c = idx & 31;
  float xr = gx[row * 96 + c],      hr = gh[row * 96 + c];
  float xz = gx[row * 96 + 32 + c], hz = gh[row * 96 + 32 + c];
  float xc = gx[row * 96 + 64 + c], hc = gh[row * 96 + 64 + c];
  float r = 1.0f / (1.0f + expf(-(xr + hr)));
  float z = 1.0f / (1.0f + expf(-(xz + hz)));
  float cand = tanhf(xc + r * hc);
  slots[idx] = (1.0f - z) * cand + z * prev[idx];
}

// decoder input (f16 NHWC): d[img][p][c] = slots[img][c] + pe_dec[p][c]
__global__ void dec_init(const float* __restrict__ slots, const float* __restrict__ pe,
                         _Float16* __restrict__ d) {
  int idx = blockIdx.x * blockDim.x + threadIdx.x;
  if (idx >= BSn * HWp * 32) return;
  int c = idx & 31;
  int rest = idx >> 5;
  int p = rest % HWp;
  int img = rest / HWp;
  d[idx] = (_Float16)(slots[img * 32 + c] + pe[p * 32 + c]);
}

// final convT: 3x3 pad1, 32->2 (flipped IOHW kernel); f16 NHWC in, f32 out
__global__ void dec4_kernel(const _Float16* __restrict__ in, const float* __restrict__ w,
                            const float* __restrict__ bias, float* __restrict__ out) {
  int idx = blockIdx.x * blockDim.x + threadIdx.x;
  if (idx >= BSn * HWp) return;
  int img = idx / HWp, p = idx - img * HWp;
  int y = p / Ww, x = p - y * Ww;
  const _Float16* ib = in + (size_t)img * HWp * 32;
  float s0 = bias[0], s1 = bias[1];
  for (int dy = 0; dy < 3; ++dy) {
    int yg = y + dy - 1;
    if (yg < 0 || yg >= Hh) continue;
    for (int dx = 0; dx < 3; ++dx) {
      int xg = x + dx - 1;
      if (xg < 0 || xg >= Ww) continue;
      int kidx = (2 - dy) * 3 + (2 - dx);
      const _Float16* px = &ib[(size_t)(yg * Ww + xg) * 32];
      for (int ci = 0; ci < 32; ++ci) {
        float v = (float)px[ci];
        s0 += v * w[(ci * 2 + 0) * 9 + kidx];
        s1 += v * w[(ci * 2 + 1) * 9 + kidx];
      }
    }
  }
  out[((size_t)img * 2 + 0) * HWp + p] = s0;
  out[((size_t)img * 2 + 1) * HWp + p] = s1;
}

// softmax over slots for mask-logits + weighted combine
__global__ void combine_kernel(const float* __restrict__ d4, float* __restrict__ out_comb,
                               float* __restrict__ out_recon, float* __restrict__ out_mask) {
  int idx = blockIdx.x * blockDim.x + threadIdx.x;       // b*HW + p
  if (idx >= Bm * HWp) return;
  int b = idx / HWp, p = idx - b * HWp;
  float rc[Ss], ml[Ss], mx = -3.4e38f;
  for (int s = 0; s < Ss; ++s) {
    size_t base = (size_t)(b * Ss + s) * 2;
    rc[s] = d4[base * HWp + p];
    ml[s] = d4[(base + 1) * HWp + p];
    mx = fmaxf(mx, ml[s]);
  }
  float sum = 0.0f;
  for (int s = 0; s < Ss; ++s) { ml[s] = expf(ml[s] - mx); sum += ml[s]; }
  float inv = 1.0f / sum, comb = 0.0f;
  for (int s = 0; s < Ss; ++s) {
    float msk = ml[s] * inv;
    comb += rc[s] * msk;
    out_recon[(size_t)(b * Ss + s) * HWp + p] = rc[s];
    out_mask[(size_t)(b * Ss + s) * HWp + p] = msk;
  }
  out_comb[idx] = comb;
}

// ---------------------------------------------------------------------------
// Host-side orchestration
// ---------------------------------------------------------------------------
static inline int cdiv(int a, int b) { return (a + b - 1) / b; }

extern "C" void kernel_launch(void* const* d_in, const int* in_sizes, int n_in,
                              void* d_out, int out_size, void* d_ws, size_t ws_size,
                              hipStream_t stream) {
  (void)in_sizes; (void)n_in; (void)out_size; (void)ws_size;
  const float* x       = (const float*)d_in[0];
  const float* noise   = (const float*)d_in[1];
  const float* enc_w1  = (const float*)d_in[2];
  const float* enc_b1  = (const float*)d_in[3];
  const float* enc_w2  = (const float*)d_in[4];
  const float* enc_b2  = (const float*)d_in[5];
  const float* enc_w3  = (const float*)d_in[6];
  const float* enc_b3  = (const float*)d_in[7];
  const float* enc_w4  = (const float*)d_in[8];
  const float* enc_b4  = (const float*)d_in[9];
  const float* pe_enc_w = (const float*)d_in[10];
  const float* pe_enc_b = (const float*)d_in[11];
  const float* ln_enc_g = (const float*)d_in[12];
  const float* ln_enc_b = (const float*)d_in[13];
  const float* mlp_w1  = (const float*)d_in[14];
  const float* mlp_b1  = (const float*)d_in[15];
  const float* mlp_w2  = (const float*)d_in[16];
  const float* mlp_b2  = (const float*)d_in[17];
  const float* slots_mu    = (const float*)d_in[18];
  const float* slots_sigma = (const float*)d_in[19];
  const float* ni_g = (const float*)d_in[20];
  const float* ni_b = (const float*)d_in[21];
  const float* ns_g = (const float*)d_in[22];
  const float* ns_b = (const float*)d_in[23];
  const float* q_w = (const float*)d_in[24];
  const float* q_b = (const float*)d_in[25];
  const float* k_w = (const float*)d_in[26];
  const float* k_b = (const float*)d_in[27];
  const float* v_w = (const float*)d_in[28];
  const float* v_b = (const float*)d_in[29];
  const float* gru_wih = (const float*)d_in[30];
  const float* gru_whh = (const float*)d_in[31];
  const float* gru_bih = (const float*)d_in[32];
  const float* gru_bhh = (const float*)d_in[33];
  const float* res_ln_g = (const float*)d_in[34];
  const float* res_ln_b = (const float*)d_in[35];
  const float* res_w1 = (const float*)d_in[36];
  const float* res_b1 = (const float*)d_in[37];
  const float* res_w2 = (const float*)d_in[38];
  const float* res_b2 = (const float*)d_in[39];
  const float* pe_dec_w = (const float*)d_in[40];
  const float* pe_dec_b = (const float*)d_in[41];
  const float* dec_w1 = (const float*)d_in[42];
  const float* dec_b1 = (const float*)d_in[43];
  const float* dec_w2 = (const float*)d_in[44];
  const float* dec_b2 = (const float*)d_in[45];
  const float* dec_w3 = (const float*)d_in[46];
  const float* dec_b3 = (const float*)d_in[47];
  const float* dec_w4 = (const float*)d_in[48];
  const float* dec_b4 = (const float*)d_in[49];

  uint8_t* base = (uint8_t*)d_ws;
  size_t off = 0;
  auto carve = [&](size_t bytes) -> uint8_t* {
    uint8_t* p = base + off;
    off += (bytes + 255) & ~(size_t)255;
    return p;
  };
  const size_t ENC_F32 = (size_t)Bm * 32 * HWp * 4;      // 26.2 MB
  const size_t ENC_F16 = (size_t)Bm * 32 * HWp * 2;      // 13.1 MB
  const size_t DEC_F16 = (size_t)BSn * 32 * HWp * 2;     // 118.0 MB
  float*    bufA  = (float*)carve(ENC_F32);
  float*    bufB  = (float*)carve(ENC_F32);
  float*    kbuf  = (float*)carve(ENC_F32);
  float*    vbuf  = (float*)carve(ENC_F32);
  _Float16* h16A  = (_Float16*)carve(ENC_F16);
  _Float16* h16B  = (_Float16*)carve(ENC_F16);
  _Float16* dh16A = (_Float16*)carve(DEC_F16);
  _Float16* dh16B = (_Float16*)carve(DEC_F16);
  float*    d4    = (float*)carve((size_t)BSn * 2 * HWp * 4);
  float*    pe_e  = (float*)carve((size_t)HWp * 32 * 4);
  float*    pe_d  = (float*)carve((size_t)HWp * 32 * 4);
  float*    dotsb = (float*)carve((size_t)Bm * Ss * HWp * 4);
  float*    attnb = (float*)carve((size_t)Bm * Ss * HWp * 4);
  float*    slots = (float*)carve(BSn * 32 * 4);
  float*    prev  = (float*)carve(BSn * 32 * 4);
  float*    snb   = (float*)carve(BSn * 32 * 4);
  float*    qb    = (float*)carve(BSn * 32 * 4);
  float*    updb  = (float*)carve(BSn * 32 * 4);
  float*    gxb   = (float*)carve(BSn * 96 * 4);
  float*    ghb   = (float*)carve(BSn * 96 * 4);
  float*    r1b   = (float*)carve(BSn * 32 * 4);
  float*    r2b   = (float*)carve(BSn * 32 * 4);
  _Float16* wp2 = (_Float16*)carve(25600 * 2);
  _Float16* wp3 = (_Float16*)carve(25600 * 2);
  _Float16* wp4 = (_Float16*)carve(25600 * 2);
  _Float16* wd1 = (_Float16*)carve(25600 * 2);
  _Float16* wd2 = (_Float16*)carve(25600 * 2);
  _Float16* wd3 = (_Float16*)carve(25600 * 2);
  _Float16* lt_m1 = (_Float16*)carve(32 * 32 * 2);
  _Float16* lt_m2 = (_Float16*)carve(32 * 32 * 2);
  _Float16* lt_k  = (_Float16*)carve(32 * 32 * 2);
  _Float16* lt_v  = (_Float16*)carve(32 * 32 * 2);
  _Float16* lt_q  = (_Float16*)carve(32 * 32 * 2);
  _Float16* lt_r1 = (_Float16*)carve(32 * 32 * 2);
  _Float16* lt_r2 = (_Float16*)carve(32 * 32 * 2);
  _Float16* lt_ih = (_Float16*)carve(96 * 32 * 2);
  _Float16* lt_hh = (_Float16*)carve(96 * 32 * 2);

  float* out = (float*)d_out;
  float* out_comb  = out;                                // 204800
  float* out_recon = out + 204800;                       // 1843200
  float* out_mask  = out + 2048000;                      // 1843200
  float* out_slots = out + 3891200;                      // 9216
  float* out_attn  = out + 3900416;                      // 1843200

  // ---- weight packing + PE tables ----
  {
    int g = cdiv(25600, 256);
    pack_conv5_frag<<<g, 256, 0, stream>>>(enc_w2, wp2, 0);
    pack_conv5_frag<<<g, 256, 0, stream>>>(enc_w3, wp3, 0);
    pack_conv5_frag<<<g, 256, 0, stream>>>(enc_w4, wp4, 0);
    pack_conv5_frag<<<g, 256, 0, stream>>>(dec_w1, wd1, 1);
    pack_conv5_frag<<<g, 256, 0, stream>>>(dec_w2, wd2, 1);
    pack_conv5_frag<<<g, 256, 0, stream>>>(dec_w3, wd3, 1);
    pack_lin_frag<<<4, 256, 0, stream>>>(mlp_w1, lt_m1, 32);
    pack_lin_frag<<<4, 256, 0, stream>>>(mlp_w2, lt_m2, 32);
    pack_lin_frag<<<4, 256, 0, stream>>>(k_w, lt_k, 32);
    pack_lin_frag<<<4, 256, 0, stream>>>(v_w, lt_v, 32);
    pack_lin_frag<<<4, 256, 0, stream>>>(q_w, lt_q, 32);
    pack_lin_frag<<<4, 256, 0, stream>>>(res_w1, lt_r1, 32);
    pack_lin_frag<<<4, 256, 0, stream>>>(res_w2, lt_r2, 32);
    pack_lin_frag<<<12, 256, 0, stream>>>(gru_wih, lt_ih, 96);
    pack_lin_frag<<<12, 256, 0, stream>>>(gru_whh, lt_hh, 96);
    pe_kernel<<<cdiv(HWp * 32, 256), 256, 0, stream>>>(pe_enc_w, pe_enc_b, pe_e);
    pe_kernel<<<cdiv(HWp * 32, 256), 256, 0, stream>>>(pe_dec_w, pe_dec_b, pe_d);
  }

  // ---- encoder ----
  conv1_kernel<<<cdiv(Bm * HWp * 32, 256), 256, 0, stream>>>(x, enc_w1, enc_b1, h16A);
  conv5_wmma<<<Bm * Hh, 320, 0, stream>>>(h16A, wp2, enc_b2, h16B, 1);
  conv5_wmma<<<Bm * Hh, 320, 0, stream>>>(h16B, wp3, enc_b3, h16A, 1);
  conv5_wmma<<<Bm * Hh, 320, 0, stream>>>(h16A, wp4, enc_b4, h16B, 1);
  enc_addpe_ln<<<Bm, 256, 0, stream>>>(h16B, pe_e, ln_enc_g, ln_enc_b, bufA);

  const int Mbig = Bm * HWp;                             // 204800 -> 12800 tiles
  const int gBig = cdiv(Mbig / 16, 8);                   // 8 waves / block
  lin32_wmma<<<gBig, 256, 0, stream>>>(bufA, lt_m1, mlp_b1, bufB, Mbig, 32, 1);
  lin32_wmma<<<gBig, 256, 0, stream>>>(bufB, lt_m2, mlp_b2, bufA, Mbig, 32, 0);
  ln_rows<<<cdiv(Mbig, 256), 256, 0, stream>>>(bufA, ni_g, ni_b, bufB, Mbig);
  lin32_wmma<<<gBig, 256, 0, stream>>>(bufB, lt_k, k_b, kbuf, Mbig, 32, 0);
  lin32_wmma<<<gBig, 256, 0, stream>>>(bufB, lt_v, v_b, vbuf, Mbig, 32, 0);

  // ---- slot attention ----
  slots_init<<<cdiv(BSn * 32, 256), 256, 0, stream>>>(slots_mu, slots_sigma, noise, slots);
  const int gSm = cdiv((BSn / 16), 8);                   // 18 tiles -> 3 blocks
  for (int it = 0; it < ITERS; ++it) {
    copyf<<<cdiv(BSn * 32, 256), 256, 0, stream>>>(slots, prev, BSn * 32);
    ln_rows<<<cdiv(BSn, 256), 256, 0, stream>>>(slots, ns_g, ns_b, snb, BSn);
    lin32_wmma<<<gSm, 256, 0, stream>>>(snb, lt_q, q_b, qb, BSn, 32, 0);
    dots_kernel<<<cdiv(Bm * HWp, 256), 256, 0, stream>>>(qb, kbuf, dotsb);
    softmax_slots<<<cdiv(Bm * HWp, 256), 256, 0, stream>>>(dotsb, attnb);
    attn_renorm<<<BSn, 256, 0, stream>>>(attnb);
    upd_kernel<<<BSn, 256, 0, stream>>>(vbuf, attnb, updb);
    lin32_wmma<<<gSm, 256, 0, stream>>>(updb, lt_ih, gru_bih, gxb, BSn, 96, 0);
    lin32_wmma<<<gSm, 256, 0, stream>>>(prev, lt_hh, gru_bhh, ghb, BSn, 96, 0);
    gru_kernel<<<cdiv(BSn * 32, 256), 256, 0, stream>>>(gxb, ghb, prev, slots);
    ln_rows<<<cdiv(BSn, 256), 256, 0, stream>>>(slots, res_ln_g, res_ln_b, r1b, BSn);
    lin32_wmma<<<gSm, 256, 0, stream>>>(r1b, lt_r1, res_b1, r2b, BSn, 32, 1);
    lin32_wmma<<<gSm, 256, 0, stream>>>(r2b, lt_r2, res_b2, r1b, BSn, 32, 0);
    addf<<<cdiv(BSn * 32, 256), 256, 0, stream>>>(slots, r1b, BSn * 32);
  }

  // ---- decoder ----
  dec_init<<<cdiv(BSn * HWp * 32, 256), 256, 0, stream>>>(slots, pe_d, dh16A);
  conv5_wmma<<<BSn * Hh, 320, 0, stream>>>(dh16A, wd1, dec_b1, dh16B, 1);
  conv5_wmma<<<BSn * Hh, 320, 0, stream>>>(dh16B, wd2, dec_b2, dh16A, 1);
  conv5_wmma<<<BSn * Hh, 320, 0, stream>>>(dh16A, wd3, dec_b3, dh16B, 1);
  dec4_kernel<<<cdiv(BSn * HWp, 256), 256, 0, stream>>>(dh16B, dec_w4, dec_b4, d4);

  // ---- outputs ----
  combine_kernel<<<cdiv(Bm * HWp, 256), 256, 0, stream>>>(d4, out_comb, out_recon, out_mask);
  copyf<<<cdiv(BSn * 32, 256), 256, 0, stream>>>(slots, out_slots, BSn * 32);
  copyf<<<cdiv(BSn * HWp, 256), 256, 0, stream>>>(attnb, out_attn, BSn * HWp);
}
